// RnnNet_54434415509795
// MI455X (gfx1250) — compile-verified, matrix operand order checked by reading, not run on hardware
//
#include <hip/hip_runtime.h>

typedef __attribute__((ext_vector_type(16))) _Float16 v16h;
typedef __attribute__((ext_vector_type(8)))  float    v8f;

namespace {
constexpr int INP      = 28;
constexpr int HID      = 64;
constexpr int NGATE    = 4 * HID;   // 256
constexpr int TSTEPS   = 28;
constexpr int OUTD     = 10;
constexpr int MTILE    = 16;        // batch rows per workgroup (= WMMA M)
constexpr int NTHREADS = 256;       // 8 wave32s
}

// A-matrix (16-bit, 16x32) element->K mapping per CDNA5 ISA 7.12.2:
// lane L: m = L%16, half = L/16; element e: K = half*8 + e (e<8), else 16 + half*8 + (e-8)
__device__ __forceinline__ int kmapA(int e, int half) {
  return (e < 8) ? (half * 8 + e) : (16 + half * 8 + (e - 8));
}

// (m, kr) -> flat index into an LDS A-fragment buffer laid out [32 lanes][16 elems],
// so a wave can load its whole A fragment as one contiguous v16h per lane.
__device__ __forceinline__ int afragIdx(int m, int kr) {
  int half, e;
  if (kr < 16) { half = kr >> 3; e = kr & 7; }
  else         { half = (kr - 16) >> 3; e = 8 + ((kr - 16) & 7); }
  return (m + 16 * half) * 16 + e;
}

// ---- fast activations: single TRANS ops (v_tanh_f32), no division chains ----
__device__ __forceinline__ float fast_tanh(float xv) {
#if __has_builtin(__builtin_amdgcn_tanhf)
  return __builtin_amdgcn_tanhf(xv);            // v_tanh_f32 (gfx1250 TRANS op)
#elif __has_builtin(__builtin_amdgcn_rcpf) && __has_builtin(__builtin_amdgcn_exp2f)
  const float e = __builtin_amdgcn_exp2f(xv * 2.885390082f);
  return 1.0f - 2.0f * __builtin_amdgcn_rcpf(1.0f + e);
#else
  const float e = __expf(2.0f * xv);
  return 1.0f - 2.0f / (1.0f + e);
#endif
}
__device__ __forceinline__ float fast_sigmoid(float xv) {
  // sigmoid(x) = 0.5 * tanh(x/2) + 0.5
  return __builtin_fmaf(0.5f, fast_tanh(0.5f * xv), 0.5f);
}

__global__ __launch_bounds__(NTHREADS) void lstm_wmma_kernel(
    const float* __restrict__ x,      // [B][28][28]
    const float* __restrict__ W_ih,   // [256][28]
    const float* __restrict__ W_hh,   // [256][64]
    const float* __restrict__ b_ih,   // [256]
    const float* __restrict__ b_hh,   // [256]
    const float* __restrict__ W_out,  // [10][64]
    const float* __restrict__ b_out,  // [10]
    float* __restrict__ out,          // [B][10]
    int batch)
{
  __shared__ float gates[MTILE][NGATE];           // 16 KB: raw gate pre-activations
  __shared__ alignas(64) _Float16 hA[2][32][16];  // 2 KB: h in A-frag layout (2 K-chunks)
  __shared__ alignas(64) _Float16 xA[32][16];     // 1 KB: x_t in A-frag layout (K padded to 32)

  const int tid  = threadIdx.x;
  const int lane = tid & 31;
  const int wave = tid >> 5;          // 0..7, owns gate N-tiles 2w and 2w+1
  const int b0   = blockIdx.x * MTILE;

  const int nloc = lane & 15;
  const int half = lane >> 4;

  // ---- build B (weight) fragments in registers, resident for all 28 steps ----
  v16h bih[2];        // W_ih^T tiles, K=0..27 zero-padded to 32
  v16h bhh[2][2];     // W_hh^T tiles, [tile][k-chunk]
  float biasv[2];
  for (int t = 0; t < 2; ++t) {
    const int n = (2 * wave + t) * 16 + nloc;   // gate column 0..255
    for (int e = 0; e < 16; ++e) {
      const int k = kmapA(e, half);
      bih[t][e] = (k < INP) ? (_Float16)W_ih[n * INP + k] : (_Float16)0.0f;
    }
    for (int c = 0; c < 2; ++c)
      for (int e = 0; e < 16; ++e) {
        const int k = c * 32 + kmapA(e, half);
        bhh[t][c][e] = (_Float16)W_hh[n * HID + k];
      }
    biasv[t] = b_ih[n] + b_hh[n];
  }

  // ---- h0 = 0 ; stage x(t=0) ----
  for (int i = tid; i < 2 * 32 * 16; i += NTHREADS)
    (&hA[0][0][0])[i] = (_Float16)0.0f;
  for (int i = tid; i < MTILE * 32; i += NTHREADS) {
    const int m = i >> 5, k = i & 31;
    float v = 0.0f;
    if (k < INP && (b0 + m) < batch) v = x[(b0 + m) * (TSTEPS * INP) + k];
    xA[0][afragIdx(m, k)] = (_Float16)v;
  }

  float cst[4] = {0.f, 0.f, 0.f, 0.f};  // per-thread cell state (4 of 1024 tile elems)

  for (int t = 0; t < TSTEPS; ++t) {
    __syncthreads();   // hA/xA staged -> visible to GEMM phase

    // ---- GEMM phase: gates = bias + x_t @ W_ih^T + h @ W_hh^T ----
    const v16h ax = *(const v16h*)&xA[lane][0];
    const v16h a0 = *(const v16h*)&hA[0][lane][0];
    const v16h a1 = *(const v16h*)&hA[1][lane][0];
    for (int tt = 0; tt < 2; ++tt) {
      v8f acc;
      for (int r = 0; r < 8; ++r) acc[r] = biasv[tt];
      acc = __builtin_amdgcn_wmma_f32_16x16x32_f16(false, ax, false, bih[tt],    (short)0, acc, false, false);
      acc = __builtin_amdgcn_wmma_f32_16x16x32_f16(false, a0, false, bhh[tt][0], (short)0, acc, false, false);
      acc = __builtin_amdgcn_wmma_f32_16x16x32_f16(false, a1, false, bhh[tt][1], (short)0, acc, false, false);
      const int n = (2 * wave + tt) * 16 + nloc;
      for (int r = 0; r < 8; ++r) gates[r + 8 * half][n] = acc[r];
    }
    __syncthreads();   // gates visible to elementwise phase

    // ---- elementwise LSTM cell update (all 256 threads, 4 elems each) ----
    for (int j = 0; j < 4; ++j) {
      const int idx = j * NTHREADS + tid;
      const int m = idx >> 6, h = idx & 63;
      const float ig = fast_sigmoid(gates[m][h]);
      const float fg = fast_sigmoid(gates[m][HID + h]);
      const float gg = fast_tanh   (gates[m][2 * HID + h]);
      const float og = fast_sigmoid(gates[m][3 * HID + h]);
      const float c  = fg * cst[j] + ig * gg;
      cst[j] = c;
      const float hv = og * fast_tanh(c);
      hA[h >> 5][0][afragIdx(m, h & 31)] = (_Float16)hv;
    }

    // ---- stage x(t+1) ----
    if (t + 1 < TSTEPS) {
      for (int i = tid; i < MTILE * 32; i += NTHREADS) {
        const int m = i >> 5, k = i & 31;
        float v = 0.0f;
        if (k < INP && (b0 + m) < batch)
          v = x[(b0 + m) * (TSTEPS * INP) + (t + 1) * INP + k];
        xA[0][afragIdx(m, k)] = (_Float16)v;
      }
    }
  }

  __syncthreads();   // final hA complete

  // ---- output head: out[b][o] = h_T . W_out[o] + b_out[o] (160 tiny dots) ----
  // Final h is read back from the swizzled f16 A-fragment buffer.
  if (tid < MTILE * OUTD) {
    const int m = tid / OUTD, o = tid % OUTD;
    const int b = b0 + m;
    if (b < batch) {
      float s = b_out[o];
      for (int h = 0; h < HID; ++h) {
        const float hv = (float)hA[h >> 5][0][afragIdx(m, h & 31)];
        s += hv * W_out[o * HID + h];
      }
      out[b * OUTD + o] = s;
    }
  }
}

extern "C" void kernel_launch(void* const* d_in, const int* in_sizes, int n_in,
                              void* d_out, int out_size, void* d_ws, size_t ws_size,
                              hipStream_t stream) {
  const float* x     = (const float*)d_in[0];
  const float* W_ih  = (const float*)d_in[1];
  const float* W_hh  = (const float*)d_in[2];
  const float* b_ih  = (const float*)d_in[3];
  const float* b_hh  = (const float*)d_in[4];
  const float* W_out = (const float*)d_in[5];
  const float* b_out = (const float*)d_in[6];
  float* out = (float*)d_out;

  const int batch   = in_sizes[0] / (INP * INP);   // 1000
  const int nblocks = (batch + MTILE - 1) / MTILE; // 63

  lstm_wmma_kernel<<<nblocks, NTHREADS, 0, stream>>>(
      x, W_ih, W_hh, b_ih, b_hh, W_out, b_out, out, batch);
}
